// MemoryGaussianMoELayer_86174223827572
// MI455X (gfx1250) — matrix operand order repaired and checked by reading, not compiled, not run on hardware
//
#include <hip/hip_runtime.h>
#include <hip/hip_bf16.h>

// ---------------------------------------------------------------------------
// MoE layer, MI455X (gfx1250): bf16 WMMA grouped-GEMM implementation.
// Sparse top-2 routing (4x FLOP cut vs dense reference), bf16 weights staged
// in workspace so the hot set (~206MB) is mostly L2-resident (192MB).
// GEMMs use 2x2 register-blocked WMMA tiles: 4 v_wmma per 8 b128 loads.
// ---------------------------------------------------------------------------

#define T_TOK 4096   // B*N
#define DDIM  1024
#define HDIM  4096
#define NEXP  8

typedef __bf16 bf16_t;
typedef __attribute__((ext_vector_type(8)))  __bf16 bf16x8;
typedef __attribute__((ext_vector_type(16))) __bf16 v16bf;
typedef __attribute__((ext_vector_type(8)))  float  v8f;

// Workspace layout (bytes). Total ~210 MB.
#define OFF_CNT   0ull
#define OFF_OFFS  64ull
#define OFF_RT    128ull                                   // int[8*4096]
#define OFF_RW    (OFF_RT + 4ull * NEXP * T_TOK)           // float[8*4096]
#define OFF_XB    (OFF_RW + 4ull * NEXP * T_TOK)           // bf16[T*D]
#define OFF_W1T   (OFF_XB + 2ull * T_TOK * DDIM)           // bf16[E*H*D]
#define OFF_W2T   (OFF_W1T + 2ull * NEXP * DDIM * HDIM)    // bf16[E*D*H]
#define OFF_HB    (OFF_W2T + 2ull * NEXP * DDIM * HDIM)    // bf16[2T*H]

__device__ __forceinline__ bf16_t f2bf(float f) {
    union { float f; unsigned int u; } v; v.f = f;
    unsigned int r = (v.u + 0x7FFFu + ((v.u >> 16) & 1u)) >> 16;
    union { unsigned short s; bf16_t b; } o; o.s = (unsigned short)r;
    return o.b;
}

// A operand: 16x32 bf16 tile, this lane's row pointer `row` (K contiguous).
// lanes 0-15: K k0..k0+15 ; lanes 16-31: K k0+16..k0+31.
__device__ __forceinline__ v16bf load_wmma_a(const bf16_t* row, int k0, int half) {
    bf16x8 lo = *(const bf16x8*)(row + k0 + half * 16);
    bf16x8 hi = *(const bf16x8*)(row + k0 + half * 16 + 8);
    return __builtin_shufflevector(lo, hi, 0,1,2,3,4,5,6,7,8,9,10,11,12,13,14,15);
}

// B operand: 32x16 bf16 tile, lane = column, column data K-contiguous in mem.
// lanes 0-15: K k0..k0+15 ; lanes 16-31: K k0+16..k0+31.
__device__ __forceinline__ v16bf load_wmma_b(const bf16_t* col, int k0, int half) {
    bf16x8 lo = *(const bf16x8*)(col + k0 + half * 16);
    bf16x8 hi = *(const bf16x8*)(col + k0 + half * 16 + 8);
    return __builtin_shufflevector(lo, hi, 0,1,2,3,4,5,6,7,8,9,10,11,12,13,14,15);
}

// ---------------------------------------------------------------------------
// 1) Transpose + fp32->bf16 convert:  in [E][R][C] f32  ->  out [E][C][R] bf16
// ---------------------------------------------------------------------------
__global__ void moe_transpose_cvt(const float* __restrict__ in, bf16_t* __restrict__ out,
                                  int R, int C) {
    __shared__ float tile[32][33];
    const int e = blockIdx.z;
    const float* src = in + (size_t)e * R * C;
    bf16_t* dst = out + (size_t)e * R * C;
    const int c = blockIdx.x * 32 + threadIdx.x;
#pragma unroll
    for (int j = 0; j < 4; ++j) {
        int r = blockIdx.y * 32 + threadIdx.y + j * 8;
        tile[threadIdx.y + j * 8][threadIdx.x] = src[(size_t)r * C + c];
    }
    __syncthreads();
    const int r2 = blockIdx.y * 32 + threadIdx.x;
#pragma unroll
    for (int j = 0; j < 4; ++j) {
        int c2 = blockIdx.x * 32 + threadIdx.y + j * 8;
        dst[(size_t)c2 * R + r2] = f2bf(tile[threadIdx.x][threadIdx.y + j * 8]);
    }
}

// ---------------------------------------------------------------------------
// 2) Gating: one wave32 per token. logits, softmax, log_softmax, top-2,
//    routing lists, and x row fp32->bf16.
// ---------------------------------------------------------------------------
__global__ void moe_gating(const float* __restrict__ x, const float* __restrict__ Wg,
                           const float* __restrict__ bg, float* __restrict__ out,
                           bf16_t* __restrict__ xb, int* __restrict__ cnt,
                           int* __restrict__ routeTok, float* __restrict__ routeW) {
    const int lane = threadIdx.x & 31;
    const int t = (blockIdx.x * blockDim.x + threadIdx.x) >> 5;
    if (t >= T_TOK) return;

    const float* xr = x + (size_t)t * DDIM;
    float acc[NEXP];
#pragma unroll
    for (int e = 0; e < NEXP; ++e) acc[e] = 0.0f;

    for (int d = lane; d < DDIM; d += 32) {
        float xv = xr[d];
        const float4* w4 = (const float4*)(Wg + (size_t)d * NEXP);
        float4 wa = w4[0], wb = w4[1];
        acc[0] += xv * wa.x; acc[1] += xv * wa.y; acc[2] += xv * wa.z; acc[3] += xv * wa.w;
        acc[4] += xv * wb.x; acc[5] += xv * wb.y; acc[6] += xv * wb.z; acc[7] += xv * wb.w;
    }
#pragma unroll
    for (int off = 16; off >= 1; off >>= 1)
#pragma unroll
        for (int e = 0; e < NEXP; ++e) acc[e] += __shfl_xor(acc[e], off, 32);
#pragma unroll
    for (int e = 0; e < NEXP; ++e) acc[e] += bg[e];

    float m = acc[0];
#pragma unroll
    for (int e = 1; e < NEXP; ++e) m = fmaxf(m, acc[e]);
    float ex[NEXP], s = 0.0f;
#pragma unroll
    for (int e = 0; e < NEXP; ++e) { ex[e] = expf(acc[e] - m); s += ex[e]; }
    const float inv_s = 1.0f / s, log_s = logf(s);
    float w[NEXP];
#pragma unroll
    for (int e = 0; e < NEXP; ++e) w[e] = ex[e] * inv_s;

    // top-2 (first index wins ties, like lax.top_k)
    int i0 = 0;
#pragma unroll
    for (int e = 1; e < NEXP; ++e) if (w[e] > w[i0]) i0 = e;
    int i1 = (i0 == 0) ? 1 : 0;
#pragma unroll
    for (int e = 0; e < NEXP; ++e) if (e != i0 && w[e] > w[i1]) i1 = e;
    const float rn = 1.0f / (w[i0] + w[i1] + 1e-9f);
    const float w0n = w[i0] * rn, w1n = w[i1] * rn;

    // x row -> bf16 staging for WMMA
    bf16_t* xbr = xb + (size_t)t * DDIM;
    for (int d = lane; d < DDIM; d += 32) xbr[d] = f2bf(xr[d]);

    if (lane == 0) {
        float* out_logp = out + (size_t)T_TOK * DDIM;
        float* out_w    = out_logp + (size_t)T_TOK * NEXP;
        float* out_idx  = out_w + (size_t)T_TOK * NEXP;
#pragma unroll
        for (int e = 0; e < NEXP; ++e) {
            out_logp[t * NEXP + e] = acc[e] - m - log_s;
            out_w[t * NEXP + e] = w[e];
        }
        out_idx[t * 2 + 0] = (float)i0;
        out_idx[t * 2 + 1] = (float)i1;
        int p0 = atomicAdd(&cnt[i0], 1);
        routeTok[i0 * T_TOK + p0] = t; routeW[i0 * T_TOK + p0] = w0n;
        int p1 = atomicAdd(&cnt[i1], 1);
        routeTok[i1 * T_TOK + p1] = t; routeW[i1 * T_TOK + p1] = w1n;
    }
}

// ---------------------------------------------------------------------------
// 3) Exclusive prefix sum of the 8 expert counts.
// ---------------------------------------------------------------------------
__global__ void moe_offsets(const int* __restrict__ cnt, int* __restrict__ offs) {
    if (threadIdx.x == 0) {
        int a = 0;
        for (int e = 0; e < NEXP; ++e) { offs[e] = a; a += cnt[e]; }
    }
}

// ---------------------------------------------------------------------------
// 4) Grouped GEMM1 + exact GELU:  h[pair, H] = gelu(x[tok] @ W1[e] + b1[e])
//    Block = 8 waves; block tile 32 rows x 256 H cols.
//    Each wave: 2x2 register-blocked WMMA (32x32 patch), K = D in steps of 32.
// ---------------------------------------------------------------------------
__global__ void __launch_bounds__(256)
moe_gemm1(const bf16_t* __restrict__ xb, const bf16_t* __restrict__ w1t,
          const float* __restrict__ b1, bf16_t* __restrict__ hbuf,
          const int* __restrict__ cnt, const int* __restrict__ offs,
          const int* __restrict__ routeTok) {
    const int e = blockIdx.z;
    const int cnt_e = cnt[e];
    const int mtile = blockIdx.y;                 // 32-row tiles
    if (mtile * 32 >= cnt_e) return;
    const int offs_e = offs[e];

    const int wave = threadIdx.x >> 5;
    const int lane = threadIdx.x & 31;
    const int half = lane >> 4;
    const int mrow = lane & 15;

    // two A row-blocks (clamped; padded rows never stored)
    int pos0 = mtile * 32 + mrow;       if (pos0 >= cnt_e) pos0 = cnt_e - 1;
    int pos1 = mtile * 32 + 16 + mrow;  if (pos1 >= cnt_e) pos1 = cnt_e - 1;
    const bf16_t* aptr0 = xb + (size_t)routeTok[e * T_TOK + pos0] * DDIM;
    const bf16_t* aptr1 = xb + (size_t)routeTok[e * T_TOK + pos1] * DDIM;

    // two B column-tiles
    const int colbase = blockIdx.x * 256 + wave * 32;
    const bf16_t* bptr0 = w1t + ((size_t)e * HDIM + colbase + mrow) * DDIM;
    const bf16_t* bptr1 = w1t + ((size_t)e * HDIM + colbase + 16 + mrow) * DDIM;

    v8f c00 = {0.f,0.f,0.f,0.f,0.f,0.f,0.f,0.f};
    v8f c01 = c00, c10 = c00, c11 = c00;
#pragma unroll 2
    for (int k0 = 0; k0 < DDIM; k0 += 32) {
        v16bf a0 = load_wmma_a(aptr0, k0, half);
        v16bf a1 = load_wmma_a(aptr1, k0, half);
        v16bf b0 = load_wmma_b(bptr0, k0, half);
        v16bf b1v = load_wmma_b(bptr1, k0, half);
        c00 = __builtin_amdgcn_wmma_f32_16x16x32_bf16(false, a0, false, b0,  (short)0, c00, false, false);
        c01 = __builtin_amdgcn_wmma_f32_16x16x32_bf16(false, a0, false, b1v, (short)0, c01, false, false);
        c10 = __builtin_amdgcn_wmma_f32_16x16x32_bf16(false, a1, false, b0,  (short)0, c10, false, false);
        c11 = __builtin_amdgcn_wmma_f32_16x16x32_bf16(false, a1, false, b1v, (short)0, c11, false, false);
    }

    const float bias0 = b1[(size_t)e * HDIM + colbase + mrow];
    const float bias1 = b1[(size_t)e * HDIM + colbase + 16 + mrow];
#pragma unroll
    for (int r = 0; r < 8; ++r) {
        const int M = r + 8 * half;
#pragma unroll
        for (int rb = 0; rb < 2; ++rb) {
            const int rowIdx = mtile * 32 + rb * 16 + M;
            if (rowIdx < cnt_e) {
                const v8f& cA = rb ? c10 : c00;
                const v8f& cB = rb ? c11 : c01;
                float v0 = cA[r] + bias0;
                float v1 = cB[r] + bias1;
                float g0 = 0.5f * v0 * (1.0f + erff(v0 * 0.70710678118654752f));
                float g1 = 0.5f * v1 * (1.0f + erff(v1 * 0.70710678118654752f));
                bf16_t* hrow = hbuf + (size_t)(offs_e + rowIdx) * HDIM;
                hrow[colbase + mrow] = f2bf(g0);
                hrow[colbase + 16 + mrow] = f2bf(g1);
            }
        }
    }
}

// ---------------------------------------------------------------------------
// 5) Grouped GEMM2 + gate-scaled scatter:  y[tok] += w * (h @ W2[e] + b2[e])
//    Block tile 32 rows x 256 D cols; wave = 2x2 WMMA block; K = H steps 32.
// ---------------------------------------------------------------------------
__global__ void __launch_bounds__(256)
moe_gemm2(const bf16_t* __restrict__ hbuf, const bf16_t* __restrict__ w2t,
          const float* __restrict__ b2, float* __restrict__ out_y,
          const int* __restrict__ cnt, const int* __restrict__ offs,
          const int* __restrict__ routeTok, const float* __restrict__ routeW) {
    const int e = blockIdx.z;
    const int cnt_e = cnt[e];
    const int mtile = blockIdx.y;
    if (mtile * 32 >= cnt_e) return;
    const int offs_e = offs[e];

    const int wave = threadIdx.x >> 5;
    const int lane = threadIdx.x & 31;
    const int half = lane >> 4;
    const int mrow = lane & 15;

    int pos0 = mtile * 32 + mrow;       if (pos0 >= cnt_e) pos0 = cnt_e - 1;
    int pos1 = mtile * 32 + 16 + mrow;  if (pos1 >= cnt_e) pos1 = cnt_e - 1;
    const bf16_t* aptr0 = hbuf + (size_t)(offs_e + pos0) * HDIM;
    const bf16_t* aptr1 = hbuf + (size_t)(offs_e + pos1) * HDIM;

    const int colbase = blockIdx.x * 256 + wave * 32;
    const bf16_t* bptr0 = w2t + ((size_t)e * DDIM + colbase + mrow) * HDIM;
    const bf16_t* bptr1 = w2t + ((size_t)e * DDIM + colbase + 16 + mrow) * HDIM;

    v8f c00 = {0.f,0.f,0.f,0.f,0.f,0.f,0.f,0.f};
    v8f c01 = c00, c10 = c00, c11 = c00;
#pragma unroll 2
    for (int k0 = 0; k0 < HDIM; k0 += 32) {
        v16bf a0 = load_wmma_a(aptr0, k0, half);
        v16bf a1 = load_wmma_a(aptr1, k0, half);
        v16bf b0 = load_wmma_b(bptr0, k0, half);
        v16bf b1v = load_wmma_b(bptr1, k0, half);
        c00 = __builtin_amdgcn_wmma_f32_16x16x32_bf16(false, a0, false, b0,  (short)0, c00, false, false);
        c01 = __builtin_amdgcn_wmma_f32_16x16x32_bf16(false, a0, false, b1v, (short)0, c01, false, false);
        c10 = __builtin_amdgcn_wmma_f32_16x16x32_bf16(false, a1, false, b0,  (short)0, c10, false, false);
        c11 = __builtin_amdgcn_wmma_f32_16x16x32_bf16(false, a1, false, b1v, (short)0, c11, false, false);
    }

    const float bias0 = b2[(size_t)e * DDIM + colbase + mrow];
    const float bias1 = b2[(size_t)e * DDIM + colbase + 16 + mrow];
#pragma unroll
    for (int r = 0; r < 8; ++r) {
        const int M = r + 8 * half;
#pragma unroll
        for (int rb = 0; rb < 2; ++rb) {
            const int rowIdx = mtile * 32 + rb * 16 + M;
            if (rowIdx < cnt_e) {
                const int tok = routeTok[e * T_TOK + rowIdx];
                const float wgt = routeW[e * T_TOK + rowIdx];
                const v8f& cA = rb ? c10 : c00;
                const v8f& cB = rb ? c11 : c01;
                float* yrow = out_y + (size_t)tok * DDIM;
                atomicAdd(yrow + colbase + mrow, (cA[r] + bias0) * wgt);
                atomicAdd(yrow + colbase + 16 + mrow, (cB[r] + bias1) * wgt);
            }
        }
    }
}

// ---------------------------------------------------------------------------
extern "C" void kernel_launch(void* const* d_in, const int* in_sizes, int n_in,
                              void* d_out, int out_size, void* d_ws, size_t ws_size,
                              hipStream_t stream) {
    const float* x  = (const float*)d_in[0];
    const float* Wg = (const float*)d_in[1];
    const float* bg = (const float*)d_in[2];
    const float* W1 = (const float*)d_in[3];
    const float* b1 = (const float*)d_in[4];
    const float* W2 = (const float*)d_in[5];
    const float* b2 = (const float*)d_in[6];
    float* out = (float*)d_out;

    char* ws = (char*)d_ws;
    int*    cnt      = (int*)(ws + OFF_CNT);
    int*    offs     = (int*)(ws + OFF_OFFS);
    int*    routeTok = (int*)(ws + OFF_RT);
    float*  routeW   = (float*)(ws + OFF_RW);
    bf16_t* xb       = (bf16_t*)(ws + OFF_XB);
    bf16_t* w1t      = (bf16_t*)(ws + OFF_W1T);
    bf16_t* w2t      = (bf16_t*)(ws + OFF_W2T);
    bf16_t* hbuf     = (bf16_t*)(ws + OFF_HB);

    hipMemsetAsync(cnt, 0, 64, stream);
    hipMemsetAsync(out, 0, (size_t)T_TOK * DDIM * sizeof(float), stream);

    // Weight convert+transpose: W1 [E,D,H] -> [E,H,D] bf16 ; W2 [E,H,D] -> [E,D,H] bf16
    moe_transpose_cvt<<<dim3(HDIM / 32, DDIM / 32, NEXP), dim3(32, 8), 0, stream>>>(W1, w1t, DDIM, HDIM);
    moe_transpose_cvt<<<dim3(DDIM / 32, HDIM / 32, NEXP), dim3(32, 8), 0, stream>>>(W2, w2t, HDIM, DDIM);

    // Gating + routing + x->bf16 (one wave per token: 4096 waves)
    moe_gating<<<T_TOK / 8, 256, 0, stream>>>(x, Wg, bg, out, xb, cnt, routeTok, routeW);
    moe_offsets<<<1, 32, 0, stream>>>(cnt, offs);

    // Grouped expert GEMMs (early-exit blocks beyond each expert's count)
    moe_gemm1<<<dim3(HDIM / 256, T_TOK / 32, NEXP), 256, 0, stream>>>(xb, w1t, b1, hbuf, cnt, offs, routeTok);
    moe_gemm2<<<dim3(DDIM / 256, T_TOK / 32, NEXP), 256, 0, stream>>>(hbuf, w2t, b2, out, cnt, offs, routeTok, routeW);
}
